// CausalityScoreModule_24043226923614
// MI455X (gfx1250) — matrix operand order repaired and checked by reading, not compiled, not run on hardware
//
#include <hip/hip_runtime.h>
#include <hip/hip_bf16.h>
#include <cstddef>

typedef _Float16 v8h  __attribute__((ext_vector_type(8)));
typedef _Float16 v16h __attribute__((ext_vector_type(16)));
typedef float    v8f  __attribute__((ext_vector_type(8)));

#define NN 4096
#define DD 128
#define KK 256

// ---------------------------------------------------------------------------
// Kernel 1: per-(b,d) column sums over N. Partial sums + atomicAdd.
// ---------------------------------------------------------------------------
__global__ void colsum_kernel(const float* __restrict__ X, float* __restrict__ colsum) {
    const int b = blockIdx.x, chunk = blockIdx.y, d = threadIdx.x;
    const float* p = X + ((size_t)b * NN + (size_t)chunk * 512) * DD + d;
    float s = 0.f;
    for (int i = 0; i < 512; ++i) s += p[(size_t)i * DD];
    atomicAdd(&colsum[b * DD + d], s);
}

// ---------------------------------------------------------------------------
// Kernel 2: one block (128 threads) per row: center, row-norm, write
// Xhat = Xc/(||Xc||+1e-6) as f16, and y = dot(X_row, W).
// ---------------------------------------------------------------------------
__global__ __launch_bounds__(128) void prep_rows(const float* __restrict__ X,
                                                 const float* __restrict__ W,
                                                 const float* __restrict__ colsum,
                                                 _Float16* __restrict__ Xhat,
                                                 float* __restrict__ y) {
    const int row = blockIdx.x;          // 0 .. B*N-1
    const int b   = row >> 12;           // N = 4096
    const int d   = threadIdx.x;
    const float invN = 1.0f / (float)NN;

    float x  = X[(size_t)row * DD + d];
    float xc = x - colsum[b * DD + d] * invN;

    __shared__ float s_ss[128], s_dt[128];
    s_ss[d] = xc * xc;
    s_dt[d] = x * W[d];
    __syncthreads();
    for (int off = 64; off > 0; off >>= 1) {
        if (d < off) { s_ss[d] += s_ss[d + off]; s_dt[d] += s_dt[d + off]; }
        __syncthreads();
    }
    float inv = 1.0f / (sqrtf(s_ss[0]) + 1e-6f);
    Xhat[(size_t)row * DD + d] = (_Float16)(xc * inv);
    if (d == 0) y[row] = s_dt[0];
}

// ---------------------------------------------------------------------------
// WMMA fragment helper (gfx1250 v_wmma_f32_16x16x32_f16 layouts, ISA 7.12.2)
// ---------------------------------------------------------------------------
__device__ inline v16h mk16(v8h lo, v8h hi) {
    return __builtin_shufflevector(lo, hi, 0,1,2,3,4,5,6,7,8,9,10,11,12,13,14,15);
}

#define WMMA_F16(A, B, C) \
    __builtin_amdgcn_wmma_f32_16x16x32_f16(false, (A), false, (B), (short)0, (C), false, false)

// ---------------------------------------------------------------------------
// Kernels 3&4: streaming |correlation| GEMM passes.
// Block = 256 threads = 8 waves; wave w owns TWO n-tiles (rows
// panel*256 + w*32 .. +32), so every B fragment read from LDS feeds two
// WMMAs (8 ds_load_b128 : 8 wmma per tile) and there are 4 independent
// accumulation chains. The 16x128 f16 m-tile is double-buffered in LDS and
// shared by all 8 waves; the global->reg load of tile mt+1 overlaps the
// WMMAs of tile mt; one barrier per iteration.
// PASS2=false: rowsum(|c|) -> d = rsqrt(rowsum)
// PASS2=true : sum(|c| * d_m * y_m) -> S = d_n * sum
// ---------------------------------------------------------------------------
template <bool PASS2>
__global__ __launch_bounds__(256) void corr_pass(const _Float16* __restrict__ Xhat,
                                                 const float* __restrict__ y,
                                                 float* __restrict__ dvec,
                                                 float* __restrict__ S) {
    const int b     = blockIdx.y;
    const int panel = blockIdx.x;            // 0..15 (256 rows per panel)
    const int wave  = threadIdx.x >> 5;
    const int lane  = threadIdx.x & 31;
    const int half  = lane >> 4;             // 0 / 1
    const int lrow  = lane & 15;
    const int n0    = panel * 256 + wave * 32;   // first tile; second at n0+16

    const _Float16* Xb = Xhat + (size_t)b * NN * DD;

    // A fragments for both n-tiles (16-bit A 16x32 layout):
    // lane<16 holds K {0-7,16-23}; lane>=16 holds K {8-15,24-31}.
    const int aoff = half * 8;
    const _Float16* arowA = Xb + (size_t)(n0 + lrow) * DD;
    const _Float16* arowB = Xb + (size_t)(n0 + 16 + lrow) * DD;
    v16h afA[4], afB[4];
#pragma unroll
    for (int kb = 0; kb < 4; ++kb) {
        afA[kb] = mk16(*(const v8h*)(arowA + kb * 32 + aoff),
                       *(const v8h*)(arowA + kb * 32 + aoff + 16));
        afB[kb] = mk16(*(const v8h*)(arowB + kb * 32 + aoff),
                       *(const v8h*)(arowB + kb * 32 + aoff + 16));
    }

    // Double-buffered m-tile: 16 rows x 128 halves, row stride padded to 136.
    __shared__ __align__(16) _Float16 tile[2][16 * 136];

    const int r   = threadIdx.x >> 4;        // staging row 0..15
    const int seg = threadIdx.x & 15;        // staging 16B segment

    // Prologue: stage tile 0.
    *(float4*)&tile[0][r * 136 + seg * 8] =
        ((const float4*)(Xb + (size_t)r * DD))[seg];
    __syncthreads();

    float acc0[8], acc1[8];
#pragma unroll
    for (int v = 0; v < 8; ++v) { acc0[v] = 0.f; acc1[v] = 0.f; }

    for (int mt = 0; mt < NN / 16; ++mt) {
        const int m0 = mt * 16;
        const bool have_nxt = (mt + 1 < NN / 16);

        // Issue the global load for tile mt+1 first (lands in regs while we compute).
        float4 nxt;
        if (have_nxt) {
            nxt = ((const float4*)(Xb + (size_t)(m0 + 16 + r) * DD))[seg];
            if (mt + 2 < NN / 16)
                __builtin_prefetch(Xb + (size_t)(m0 + 32 + r) * DD + seg * 8, 0, 1);
        }

        // Pass-2 column weight, loaded ahead of the WMMAs.
        float g = 1.0f;
        if (PASS2) {
            const int m = m0 + lrow;
            g = dvec[(size_t)b * NN + m] * y[(size_t)b * NN + m];
        }

        // Load all 4 B fragments up front (one dscnt wait covers them).
        const _Float16* tb = tile[mt & 1];
        v16h bf[4];
#pragma unroll
        for (int kb = 0; kb < 4; ++kb) {
            const _Float16* bp = &tb[lrow * 136 + kb * 32 + half * 8];
            bf[kb] = mk16(*(const v8h*)bp, *(const v8h*)(bp + 16));
        }

        // Four independent accumulation chains (2 n-tiles x 2 k-halves).
        v8f c00 = {0.f,0.f,0.f,0.f,0.f,0.f,0.f,0.f};
        v8f c01 = {0.f,0.f,0.f,0.f,0.f,0.f,0.f,0.f};
        v8f c10 = {0.f,0.f,0.f,0.f,0.f,0.f,0.f,0.f};
        v8f c11 = {0.f,0.f,0.f,0.f,0.f,0.f,0.f,0.f};
        c00 = WMMA_F16(afA[0], bf[0], c00);
        c10 = WMMA_F16(afB[0], bf[0], c10);
        c01 = WMMA_F16(afA[2], bf[2], c01);
        c11 = WMMA_F16(afB[2], bf[2], c11);
        c00 = WMMA_F16(afA[1], bf[1], c00);
        c10 = WMMA_F16(afB[1], bf[1], c10);
        c01 = WMMA_F16(afA[3], bf[3], c01);
        c11 = WMMA_F16(afB[3], bf[3], c11);

#pragma unroll
        for (int v = 0; v < 8; ++v) {
            const float a0 = fabsf(c00[v] + c01[v]);
            const float a1 = fabsf(c10[v] + c11[v]);
            acc0[v] += PASS2 ? a0 * g : a0;
            acc1[v] += PASS2 ? a1 * g : a1;
        }

        // Store the prefetched tile into the alternate buffer, then one barrier.
        if (have_nxt)
            *(float4*)&tile[(mt + 1) & 1][r * 136 + seg * 8] = nxt;
        __syncthreads();
    }

    // Reduce over the 16 columns held by the lanes of each half-wave.
#pragma unroll
    for (int v = 0; v < 8; ++v) {
        float t0 = acc0[v], t1 = acc1[v];
        t0 += __shfl_xor(t0, 8, 32);  t1 += __shfl_xor(t1, 8, 32);
        t0 += __shfl_xor(t0, 4, 32);  t1 += __shfl_xor(t1, 4, 32);
        t0 += __shfl_xor(t0, 2, 32);  t1 += __shfl_xor(t1, 2, 32);
        t0 += __shfl_xor(t0, 1, 32);  t1 += __shfl_xor(t1, 1, 32);
        acc0[v] = t0; acc1[v] = t1;
    }
    if (lrow == 0) {
#pragma unroll
        for (int v = 0; v < 8; ++v) {
            const int rowA = n0 + half * 8 + v;        // C/D: M = v + 8*half
            const int rowB = rowA + 16;
            if (!PASS2) {
                dvec[(size_t)b * NN + rowA] = rsqrtf(acc0[v]);
                dvec[(size_t)b * NN + rowB] = rsqrtf(acc1[v]);
            } else {
                S[(size_t)b * NN + rowA] = dvec[(size_t)b * NN + rowA] * acc0[v];
                S[(size_t)b * NN + rowB] = dvec[(size_t)b * NN + rowB] * acc1[v];
            }
        }
    }
}

// ---------------------------------------------------------------------------
// Kernel 5: per-batch bitonic top-K (value desc, index asc tiebreak — matches
// lax.top_k), gather F_c in score order, and the two means.
// ---------------------------------------------------------------------------
__global__ __launch_bounds__(1024) void topk_kernel(const float* __restrict__ S,
                                                    const float* __restrict__ X,
                                                    const float* __restrict__ colsum,
                                                    float* __restrict__ Fc,
                                                    float* __restrict__ meanC,
                                                    float* __restrict__ meanNC) {
    const int b = blockIdx.x, t = threadIdx.x;
    __shared__ float key[NN];
    __shared__ int   sidx[NN];

    for (int i = t; i < NN; i += 1024) { key[i] = S[(size_t)b * NN + i]; sidx[i] = i; }
    __syncthreads();

    for (int size = 2; size <= NN; size <<= 1) {
        for (int stride = size >> 1; stride > 0; stride >>= 1) {
            for (int p = t; p < NN / 2; p += 1024) {
                const int i = ((p / stride) * stride * 2) + (p % stride);
                const int j = i + stride;
                const bool dirDesc = ((i & size) == 0);
                float ki = key[i], kj = key[j];
                int   ii = sidx[i], ij = sidx[j];
                const bool before = (ki > kj) || (ki == kj && ii < ij);
                if (before != dirDesc) {
                    key[i] = kj; key[j] = ki;
                    sidx[i] = ij; sidx[j] = ii;
                }
            }
            __syncthreads();
        }
    }

    // Gather F_c (descending-score order) + column partial sums.
    const int r = t >> 7, d = t & 127;
    float local = 0.f;
    for (int k = r; k < KK; k += 8) {
        const int idx = sidx[k];
        const float v = X[((size_t)b * NN + idx) * DD + d];
        Fc[((size_t)b * KK + k) * DD + d] = v;
        local += v;
    }
    float* part = key;                 // keys no longer needed: reuse LDS
    part[r * 128 + d] = local;
    __syncthreads();
    if (t < 128) {
        float s = 0.f;
        for (int rr = 0; rr < 8; ++rr) s += part[rr * 128 + t];
        const float total = colsum[b * DD + t];      // sum over all N rows
        meanC[b * DD + t]  = s / (float)KK;
        meanNC[b * DD + t] = (total - s) / (float)(NN - KK);
    }
}

// ---------------------------------------------------------------------------
extern "C" void kernel_launch(void* const* d_in, const int* in_sizes, int n_in,
                              void* d_out, int out_size, void* d_ws, size_t ws_size,
                              hipStream_t stream) {
    (void)n_in; (void)out_size; (void)ws_size;
    const int B = in_sizes[0] / (NN * DD);           // 8

    const float* X = (const float*)d_in[0];
    const float* W = (const float*)d_in[1];

    float* out    = (float*)d_out;
    float* Sout   = out;                             // [B,N]
    float* Fc     = Sout + (size_t)B * NN;           // [B,K,D]
    float* meanC  = Fc + (size_t)B * KK * DD;        // [B,1,D]
    float* meanNC = meanC + (size_t)B * DD;          // [B,1,D]

    char* ws = (char*)d_ws;
    size_t off = 0;
    float* colsum = (float*)(ws + off); off += (size_t)B * DD * sizeof(float);
    float* yvec   = (float*)(ws + off); off += (size_t)B * NN * sizeof(float);
    float* dvec   = (float*)(ws + off); off += (size_t)B * NN * sizeof(float);
    off = (off + 255) & ~(size_t)255;
    _Float16* Xhat = (_Float16*)(ws + off);          // [B,N,D] f16 (8 MB)

    hipMemsetAsync(colsum, 0, (size_t)B * DD * sizeof(float), stream);
    colsum_kernel<<<dim3(B, 8), 128, 0, stream>>>(X, colsum);
    prep_rows<<<B * NN, 128, 0, stream>>>(X, W, colsum, Xhat, yvec);
    corr_pass<false><<<dim3(NN / 256, B), 256, 0, stream>>>(Xhat, yvec, dvec, Sout);
    corr_pass<true ><<<dim3(NN / 256, B), 256, 0, stream>>>(Xhat, yvec, dvec, Sout);
    topk_kernel<<<B, 1024, 0, stream>>>(Sout, X, colsum, Fc, meanC, meanNC);
}
